// ClassifyModelMOE_53772990546246
// MI455X (gfx1250) — matrix operand order repaired and checked by reading, not compiled, not run on hardware
//
#include <hip/hip_runtime.h>
#include <stdint.h>

#define BATCH 8192
#define NE    5
#define TOPK  3
#define DIN   3200
#define DH    128
#define NCLS  10

typedef __bf16    bf16_t;
typedef bf16_t    v16bf __attribute__((ext_vector_type(16)));
typedef float     v8f   __attribute__((ext_vector_type(8)));
typedef uint32_t  u32x4 __attribute__((ext_vector_type(4)));

union Frag { u32x4 q[2]; v16bf v; };

__device__ __forceinline__ uint16_t f2bf(float f) {
    uint32_t u = __float_as_uint(f);
    u += 0x7FFFu + ((u >> 16) & 1u);      // round-to-nearest-even
    return (uint16_t)(u >> 16);
}
__device__ __forceinline__ float bf2f(uint16_t h) {
    return __uint_as_float(((uint32_t)h) << 16);
}

// Async copy 16 bytes global -> LDS (gfx1250 GLOBAL_LOAD_ASYNC_TO_LDS_B128,
// tracked by ASYNCcnt).  laddr = LDS byte address, gaddr = 64-bit global addr.
__device__ __forceinline__ void async_g2l_b128(uint32_t laddr, uint64_t gaddr) {
    asm volatile("global_load_async_to_lds_b128 %0, %1, off"
                 :: "v"(laddr), "v"(gaddr) : "memory");
}
__device__ __forceinline__ void wait_asynccnt0() {
    asm volatile("s_wait_asynccnt 0x0" ::: "memory");
}

// ---------------------------------------------------------------------------
// Kernel 1: conv1 (5x5 valid) + ReLU + 2x2 maxpool.  x[B,1,28,28] -> h1[B,16,12,12]
// ---------------------------------------------------------------------------
__global__ __launch_bounds__(256) void k_conv1_pool(
    const float* __restrict__ x, const float* __restrict__ w,
    const float* __restrict__ bias, float* __restrict__ h1)
{
    int idx = blockIdx.x * 256 + threadIdx.x;
    if (idx >= BATCH * 16 * 12 * 12) return;
    int j = idx % 12; int t = idx / 12;
    int i = t % 12;   t /= 12;
    int c = t % 16;   int b = t / 16;
    const float* xb = x + (size_t)b * 28 * 28;
    const float* wc = w + c * 25;
    float bc = bias[c];
    float m = -3.4e38f;
    #pragma unroll
    for (int dy = 0; dy < 2; ++dy)
    #pragma unroll
    for (int dx = 0; dx < 2; ++dx) {
        int oy = 2 * i + dy, ox = 2 * j + dx;
        float s = bc;
        #pragma unroll
        for (int ky = 0; ky < 5; ++ky) {
            const float* row = xb + (oy + ky) * 28 + ox;
            const float* wr  = wc + ky * 5;
            #pragma unroll
            for (int kx = 0; kx < 5; ++kx) s = fmaf(row[kx], wr[kx], s);
        }
        s = fmaxf(s, 0.f);
        m = fmaxf(m, s);
    }
    h1[idx] = m;
}

// ---------------------------------------------------------------------------
// Kernel 2: conv2 (3x3 valid) + ReLU, per-image block, image in LDS.
// h1[B,16,12,12] -> feat bf16 [B,3200]
// ---------------------------------------------------------------------------
__global__ __launch_bounds__(256) void k_conv2(
    const float* __restrict__ h1, const float* __restrict__ w2,
    const float* __restrict__ b2, uint16_t* __restrict__ featbf)
{
    __shared__ float sh[16 * 12 * 12];
    int b = blockIdx.x;
    const float* src = h1 + (size_t)b * 2304;
    for (int i = threadIdx.x; i < 2304; i += 256) sh[i] = src[i];
    __syncthreads();
    for (int o = threadIdx.x; o < 3200; o += 256) {
        int c = o / 100, p = o % 100;
        int y = p / 10,  xx = p % 10;
        float s = b2[c];
        const float* wc = w2 + c * 16 * 9;
        for (int ci = 0; ci < 16; ++ci) {
            const float* sp = sh + ci * 144 + y * 12 + xx;
            const float* wp = wc + ci * 9;
            #pragma unroll
            for (int ky = 0; ky < 3; ++ky)
            #pragma unroll
            for (int kx = 0; kx < 3; ++kx)
                s = fmaf(sp[ky * 12 + kx], wp[ky * 3 + kx], s);
        }
        featbf[(size_t)b * DIN + o] = f2bf(fmaxf(s, 0.f));
    }
}

// ---------------------------------------------------------------------------
// Kernel 3: pack expert weights to column-major bf16 for B-fragment loads.
// ---------------------------------------------------------------------------
__global__ __launch_bounds__(256) void k_pack(
    const float* __restrict__ W1e, const float* __restrict__ W2e,
    uint16_t* __restrict__ W1T, uint16_t* __restrict__ W2T)
{
    int idx = blockIdx.x * 256 + threadIdx.x;
    const int n1 = NE * DH * DIN;   // 2,048,000
    if (idx < n1) {
        int k = idx % DIN; int t = idx / DIN;
        int n = t % DH;    int e = t / DH;
        W1T[idx] = f2bf(W1e[((size_t)e * DIN + k) * DH + n]);
    }
    const int n2 = NE * DH * DH;    // 81,920
    if (idx < n2) {
        int k = idx % DH; int t = idx / DH;
        int n = t % DH;   int e = t / DH;
        W2T[idx] = f2bf(W2e[((size_t)e * DH + k) * DH + n]);
    }
}

// ---------------------------------------------------------------------------
// Kernel 4: router gate.  One wave32 per row.  weights[B,5] f32.
// ---------------------------------------------------------------------------
__global__ __launch_bounds__(256) void k_gate(
    const uint16_t* __restrict__ featbf,
    const float* __restrict__ Wg,  const float* __restrict__ bg,
    const float* __restrict__ Wgl, const float* __restrict__ bgl,
    float* __restrict__ wexp)
{
    int gtid = blockIdx.x * 256 + threadIdx.x;
    int row  = gtid >> 5;
    int lane = gtid & 31;
    const uint16_t* fr = featbf + (size_t)row * DIN;
    float acc[NE]  = {0.f, 0.f, 0.f, 0.f, 0.f};
    float accg[NE] = {0.f, 0.f, 0.f, 0.f, 0.f};
    for (int k = lane; k < DIN; k += 32) {
        float f = bf2f(fr[k]);
        const float* g  = Wg  + k * NE;
        const float* gl = Wgl + k * NE;
        #pragma unroll
        for (int j = 0; j < NE; ++j) {
            acc[j]  = fmaf(f, g[j],  acc[j]);
            accg[j] = fmaf(f, gl[j], accg[j]);
        }
    }
    #pragma unroll
    for (int off = 16; off >= 1; off >>= 1) {
        #pragma unroll
        for (int j = 0; j < NE; ++j) {
            acc[j]  += __shfl_xor(acc[j],  off);
            accg[j] += __shfl_xor(accg[j], off);
        }
    }
    if (lane == 0) {
        float gate[NE];
        #pragma unroll
        for (int j = 0; j < NE; ++j) {
            float h   = acc[j] + bg[j];
            float glu = 1.f / (1.f + __expf(-(accg[j] + bgl[j])));
            gate[j] = h * glu;
        }
        int used = 0; float tv[TOPK]; int ti[TOPK];
        #pragma unroll
        for (int t = 0; t < TOPK; ++t) {
            float bv = -3.4e38f; int bi = 0;
            #pragma unroll
            for (int j = 0; j < NE; ++j)
                if (!((used >> j) & 1) && gate[j] > bv) { bv = gate[j]; bi = j; }
            used |= (1 << bi); tv[t] = bv; ti[t] = bi;
        }
        float mx = tv[0];
        float e0 = __expf(tv[0] - mx), e1 = __expf(tv[1] - mx), e2 = __expf(tv[2] - mx);
        float inv = 1.f / (e0 + e1 + e2);
        float w[NE] = {0.f, 0.f, 0.f, 0.f, 0.f};
        w[ti[0]] = e0 * inv; w[ti[1]] = e1 * inv; w[ti[2]] = e2 * inv;
        #pragma unroll
        for (int j = 0; j < NE; ++j) wexp[(size_t)row * NE + j] = w[j];
    }
}

// ---------------------------------------------------------------------------
// Kernel 5: GEMM1 (WMMA bf16): e1 = tanh(feat @ W1 + b1).
// M=8192, K=3200, N=640.  Block = 8 waves, tile 128x64 (4 wmma/wave/K-step).
// B tile (32x64 bf16 = 4KB) staged via async global->LDS, double-buffered.
// ---------------------------------------------------------------------------
__global__ __launch_bounds__(256) void k_gemm1(
    const uint16_t* __restrict__ featbf, const uint16_t* __restrict__ W1T,
    const float* __restrict__ b1e, uint16_t* __restrict__ e1)
{
    __shared__ __align__(16) uint16_t ldsB[2][2048];   // 2 x (64 cols x 32 K) bf16

    const int tid  = threadIdx.x;
    const int wv   = tid >> 5;
    const int lane = tid & 31;
    const int lm   = lane & 15;
    const int hi   = lane >> 4;
    const int m0   = blockIdx.x * 128 + wv * 16;
    const int nG   = blockIdx.y * 64;               // 64-column block (expert-aligned)

    // A pointer: row (m0+lm), K offset hi*16 / +8 (contiguous K order, matching B).
    const uint16_t* Aptr = featbf + (size_t)(m0 + lm) * DIN;
    const int aK = hi * 16;

    // Async B loader: thread t moves 16 bytes: column bn = t>>2, segment bseg = t&3.
    const int bn   = tid >> 2;
    const int bseg = tid & 3;
    const uint16_t* gB = W1T + (size_t)(nG + bn) * DIN + bseg * 8;
    const uint32_t ldsBase = (uint32_t)(uintptr_t)(&ldsB[0][0]);
    const uint32_t myLds   = ldsBase + (uint32_t)(bn * 64 + bseg * 16);

    v8f acc0 = {}, acc1 = {}, acc2 = {}, acc3 = {};

    // prologue: fetch tile 0 into buffer 0
    async_g2l_b128(myLds, (uint64_t)(uintptr_t)gB);

    int buf = 0;
    for (int kb = 0; kb < DIN; kb += 32) {
        wait_asynccnt0();           // my 16B of tile (kb) has landed
        __syncthreads();            // everyone's 16B landed; prev compute done
        if (kb + 32 < DIN) {        // uniform branch
            async_g2l_b128(myLds + (uint32_t)((buf ^ 1) * 4096),
                           (uint64_t)(uintptr_t)(gB + kb + 32));
            __builtin_prefetch(Aptr + kb + 32, 0, 3);
        }
        Frag a;
        a.q[0] = *(const u32x4*)(Aptr + kb + aK);        // K = kb+aK   .. +7
        a.q[1] = *(const u32x4*)(Aptr + kb + aK + 8);    // K = kb+aK+8 .. +15
        const uint16_t* bbuf = &ldsB[buf][0];
        #pragma unroll
        for (int j = 0; j < 4; ++j) {
            Frag b;
            const uint16_t* bcol = bbuf + (j * 16 + lm) * 32 + hi * 16;
            b.q[0] = *(const u32x4*)(bcol);              // K = kb+16*hi .. +15
            b.q[1] = *(const u32x4*)(bcol + 8);
            v8f& acc = (j == 0) ? acc0 : (j == 1) ? acc1 : (j == 2) ? acc2 : acc3;
            acc = __builtin_amdgcn_wmma_f32_16x16x32_bf16(
                      false, a.v, false, b.v, (short)0, acc, false, false);
        }
        buf ^= 1;
    }

    // epilogue: tanh(acc + bias) -> e1[e][m][nIn] bf16
    const int e = nG / DH;                  // uniform within block (64 | 128)
    #pragma unroll
    for (int j = 0; j < 4; ++j) {
        const v8f& acc = (j == 0) ? acc0 : (j == 1) ? acc1 : (j == 2) ? acc2 : acc3;
        const int nIn  = (nG % DH) + j * 16 + lm;
        const float bias = b1e[e * DH + nIn];
        #pragma unroll
        for (int r = 0; r < 8; ++r) {
            int m = m0 + r + hi * 8;
            float v = tanhf(acc[r] + bias);
            e1[((size_t)e * BATCH + m) * DH + nIn] = f2bf(v);
        }
    }
}

// ---------------------------------------------------------------------------
// Kernel 6: GEMM2 (WMMA bf16) + gated combine:
// moe[b,g] = sum_e w[b,e] * tanh(e1[e,b,:] @ W2[e] + b2[e])
// ---------------------------------------------------------------------------
__global__ __launch_bounds__(256) void k_gemm2(
    const uint16_t* __restrict__ e1, const uint16_t* __restrict__ W2T,
    const float* __restrict__ b2e, const float* __restrict__ wexp,
    float* __restrict__ moe)
{
    const int tid  = threadIdx.x;
    const int wv   = tid >> 5;
    const int lane = tid & 31;
    const int lm   = lane & 15;
    const int hi   = lane >> 4;
    const int m0   = blockIdx.x * 128 + wv * 16;
    const int n0   = blockIdx.y * 16;

    float out[8] = {0.f, 0.f, 0.f, 0.f, 0.f, 0.f, 0.f, 0.f};
    for (int e = 0; e < NE; ++e) {
        const uint16_t* Ab = e1  + ((size_t)e * BATCH + m0 + lm) * DH;
        const uint16_t* Bb = W2T + ((size_t)e * DH + n0 + lm) * DH;
        v8f acc = {};
        #pragma unroll
        for (int kb = 0; kb < DH; kb += 32) {
            Frag a, b;
            a.q[0] = *(const u32x4*)(Ab + kb + hi * 16);
            a.q[1] = *(const u32x4*)(Ab + kb + hi * 16 + 8);
            b.q[0] = *(const u32x4*)(Bb + kb + hi * 16);
            b.q[1] = *(const u32x4*)(Bb + kb + hi * 16 + 8);
            acc = __builtin_amdgcn_wmma_f32_16x16x32_bf16(
                      false, a.v, false, b.v, (short)0, acc, false, false);
        }
        const float bias = b2e[e * DH + n0 + lm];
        #pragma unroll
        for (int r = 0; r < 8; ++r) {
            int m = m0 + r + hi * 8;
            float v = tanhf(acc[r] + bias);
            out[r] = fmaf(wexp[(size_t)m * NE + e], v, out[r]);
        }
    }
    #pragma unroll
    for (int r = 0; r < 8; ++r) {
        int m = m0 + r + hi * 8;
        moe[(size_t)m * DH + n0 + lm] = out[r];
    }
}

// ---------------------------------------------------------------------------
// Kernel 7: final classifier + softmax.  One thread per row.
// ---------------------------------------------------------------------------
__global__ __launch_bounds__(256) void k_final(
    const float* __restrict__ moe, const float* __restrict__ Ws,
    const float* __restrict__ bs, float* __restrict__ out)
{
    int b = blockIdx.x * 256 + threadIdx.x;
    if (b >= BATCH) return;
    const float* r = moe + (size_t)b * DH;
    float lg[NCLS];
    #pragma unroll
    for (int j = 0; j < NCLS; ++j) lg[j] = bs[j];
    for (int k = 0; k < DH; ++k) {
        float f = r[k];
        const float* w = Ws + k * NCLS;
        #pragma unroll
        for (int j = 0; j < NCLS; ++j) lg[j] = fmaf(f, w[j], lg[j]);
    }
    float mx = lg[0];
    #pragma unroll
    for (int j = 1; j < NCLS; ++j) mx = fmaxf(mx, lg[j]);
    float s = 0.f;
    #pragma unroll
    for (int j = 0; j < NCLS; ++j) { lg[j] = __expf(lg[j] - mx); s += lg[j]; }
    float inv = 1.f / s;
    #pragma unroll
    for (int j = 0; j < NCLS; ++j) out[(size_t)b * NCLS + j] = lg[j] * inv;
}

// ---------------------------------------------------------------------------
extern "C" void kernel_launch(void* const* d_in, const int* in_sizes, int n_in,
                              void* d_out, int out_size, void* d_ws, size_t ws_size,
                              hipStream_t stream)
{
    const float* x    = (const float*)d_in[0];
    const float* c1w  = (const float*)d_in[1];
    const float* c1b  = (const float*)d_in[2];
    const float* c2w  = (const float*)d_in[3];
    const float* c2b  = (const float*)d_in[4];
    const float* Wg   = (const float*)d_in[5];
    const float* bg   = (const float*)d_in[6];
    const float* Wglu = (const float*)d_in[7];
    const float* bglu = (const float*)d_in[8];
    const float* W1e  = (const float*)d_in[9];
    const float* b1e  = (const float*)d_in[10];
    const float* W2e  = (const float*)d_in[11];
    const float* b2e  = (const float*)d_in[12];
    const float* Ws   = (const float*)d_in[13];
    const float* bs   = (const float*)d_in[14];
    float* out = (float*)d_out;

    char* ws = (char*)d_ws;
    size_t off = 0;
    auto take = [&](size_t bytes) -> char* {
        char* p = ws + off;
        off = (off + bytes + 255) & ~(size_t)255;
        return p;
    };
    uint16_t* featbf = (uint16_t*)take((size_t)BATCH * DIN * 2);        // 52.4 MB
    uint16_t* W1T    = (uint16_t*)take((size_t)NE * DH * DIN * 2);      // 4.1 MB
    uint16_t* W2T    = (uint16_t*)take((size_t)NE * DH * DH * 2);       // 0.16 MB
    uint16_t* e1buf  = (uint16_t*)take((size_t)NE * BATCH * DH * 2);    // 10.5 MB
    float*    wexp   = (float*)take((size_t)BATCH * NE * 4);            // 0.16 MB
    float*    moe    = (float*)take((size_t)BATCH * DH * 4);            // 4.2 MB
    float*    h1     = (float*)take((size_t)BATCH * 16 * 12 * 12 * 4);  // 75.5 MB

    k_conv1_pool<<<(BATCH * 16 * 144) / 256, 256, 0, stream>>>(x, c1w, c1b, h1);
    k_pack<<<(NE * DH * DIN) / 256, 256, 0, stream>>>(W1e, W2e, W1T, W2T);
    k_conv2<<<BATCH, 256, 0, stream>>>(h1, c2w, c2b, featbf);
    k_gate<<<(BATCH * 32) / 256, 256, 0, stream>>>(featbf, Wg, bg, Wglu, bglu, wexp);
    dim3 g1(BATCH / 128, (NE * DH) / 64);
    k_gemm1<<<g1, 256, 0, stream>>>(featbf, W1T, b1e, e1buf);
    dim3 g2(BATCH / 128, DH / 16);
    k_gemm2<<<g2, 256, 0, stream>>>(e1buf, W2T, b2e, wexp, moe);
    k_final<<<BATCH / 256, 256, 0, stream>>>(moe, Ws, bs, out);
}
